// QuantizedLinear_9380208575266
// MI455X (gfx1250) — compile-verified, matrix operand order checked by reading, not run on hardware
//
#include <hip/hip_runtime.h>

typedef __attribute__((ext_vector_type(16))) __bf16 v16bf;
typedef __attribute__((ext_vector_type(8)))  __bf16 v8bf;
typedef __attribute__((ext_vector_type(4)))  __bf16 v4bf;
typedef __attribute__((ext_vector_type(8)))  float  v8f;
typedef __attribute__((ext_vector_type(4)))  float  v4f;
typedef __attribute__((ext_vector_type(4)))  int    v4i;

#define IN_F   2048
#define OUT_F  8192
#define M_TOT  4096

// ---- phase-2 GEMM tiling ----
#define BM     128
#define BN     256
#define BK     32
#define LDSTR  40                   // elements; 80B row stride (5x16B, conflict-free)
#define XBYTES (BM * LDSTR * 2)     // 10240
#define WBYTES (BN * LDSTR * 2)     // 20480
#define STAGE  (XBYTES + WBYTES)    // 30720
#define KT     (IN_F / BK)          // 64

__device__ const float NF4TAB[16] = {
    -1.0f, -0.6961928009986877f, -0.5250730514526367f, -0.39491748809814453f,
    -0.28444138169288635f, -0.18477343022823334f, -0.09105003625154495f, 0.0f,
    0.07958029955625534f, 0.16093020141124725f, 0.24611230194568634f,
    0.33791524171829224f, 0.44070982933044434f, 0.5626170039176941f,
    0.7229568362236023f, 1.0f};

__device__ __forceinline__ __bf16 f2bf(float f) {
  unsigned u = __builtin_bit_cast(unsigned, f);
  unsigned r = u + 0x7FFFu + ((u >> 16) & 1u);
  unsigned short h = (unsigned short)(r >> 16);
  return __builtin_bit_cast(__bf16, h);
}

__device__ __forceinline__ void async_b128(unsigned ldsoff, const void* gptr) {
  // memory -> LDS, no VGPR staging; tracked by ASYNCcnt
  asm volatile("global_load_async_to_lds_b128 %0, %1, off"
               :: "v"(ldsoff), "v"(gptr) : "memory");
}

// ================= phase 1a: NF4 dequant W -> bf16 =================
__global__ __launch_bounds__(256)
void dequant_w(const int* __restrict__ Q, const float* __restrict__ S,
               __bf16* __restrict__ Wb) {
  __shared__ float nf4[16];
  if (threadIdx.x < 16) nf4[threadIdx.x] = NF4TAB[threadIdx.x];
  __syncthreads();
  size_t base = ((size_t)blockIdx.x * 256 + threadIdx.x) * 8;  // 8 codes/thread
  v4i q0 = *(const v4i*)(Q + base);
  v4i q1 = *(const v4i*)(Q + base + 4);
  float s = S[base >> 6];            // 8 codes never straddle a 64-block
  v8bf o;
#pragma unroll
  for (int j = 0; j < 4; ++j) o[j]     = f2bf(nf4[q0[j] & 15] * s);
#pragma unroll
  for (int j = 0; j < 4; ++j) o[j + 4] = f2bf(nf4[q1[j] & 15] * s);
  *(v8bf*)(Wb + base) = o;
}

// ================= phase 1b: X f32 -> bf16 =================
__global__ __launch_bounds__(256)
void convert_x(const float* __restrict__ X, __bf16* __restrict__ Xb) {
  size_t base = ((size_t)blockIdx.x * 256 + threadIdx.x) * 8;
  v4f x0 = *(const v4f*)(X + base);
  v4f x1 = *(const v4f*)(X + base + 4);
  v8bf o;
#pragma unroll
  for (int j = 0; j < 4; ++j) o[j]     = f2bf(x0[j]);
#pragma unroll
  for (int j = 0; j < 4; ++j) o[j + 4] = f2bf(x1[j]);
  *(v8bf*)(Xb + base) = o;
}

// ================= phase 2: bf16 WMMA GEMM, async double-buffer =================
__global__ __launch_bounds__(256, 1)
void bf16_wmma_gemm(const __bf16* __restrict__ Xb, const __bf16* __restrict__ Wb,
                    const float* __restrict__ Bv, float* __restrict__ Y) {
  extern __shared__ char smem[];
  const unsigned lds0 = (unsigned)(size_t)(void*)smem;  // dynamic-LDS byte base

  const int tid  = threadIdx.x;
  const int lane = tid & 31;
  const int wave = tid >> 5;
  const int wm   = wave >> 2;    // 0..1  -> 64-row slab
  const int wn   = wave & 3;     // 0..3  -> 64-col slab
  const int half = lane >> 4;
  const int r16  = lane & 15;

  const int m0 = blockIdx.y * BM;
  const int n0 = blockIdx.x * BN;

  // ---- per-thread async chunk addresses (6 b128 chunks / thread / stage) ----
  // X tile: 128 rows x 4 chunks = 512 ; W tile: 256 rows x 4 chunks = 1024
  const char* gbase[6];
  unsigned    lbase[6];
#pragma unroll
  for (int i = 0; i < 6; ++i) {
    int g = i * 256 + tid;
    if (g < 512) {                       // i = 0,1 : X tile
      int row = g >> 2, ch = g & 3;
      gbase[i] = (const char*)(Xb + (size_t)(m0 + row) * IN_F + ch * 8);
      lbase[i] = lds0 + (unsigned)(row * 80 + ch * 16);
    } else {                             // i = 2..5 : W tile
      int gw = g - 512;
      int row = gw >> 2, ch = gw & 3;
      gbase[i] = (const char*)(Wb + (size_t)(n0 + row) * IN_F + ch * 8);
      lbase[i] = lds0 + (unsigned)(XBYTES + row * 80 + ch * 16);
    }
  }

  v8f acc[4][4] = {};

  // prologue: stage 0 in flight
#pragma unroll
  for (int i = 0; i < 6; ++i) async_b128(lbase[i], gbase[i]);

  for (int kt = 0; kt < KT; ++kt) {
    const int cur = kt & 1;
    if (kt + 1 < KT) {
      const unsigned soff = (cur ^ 1) * STAGE;
      const size_t   goff = (size_t)(kt + 1) * (BK * 2);   // 64B per K-step
#pragma unroll
      for (int i = 0; i < 6; ++i) async_b128(lbase[i] + soff, gbase[i] + goff);
      asm volatile("s_wait_asynccnt 6" ::: "memory");      // stage `cur` landed
    } else {
      asm volatile("s_wait_asynccnt 0" ::: "memory");
    }
    __syncthreads();

    const char* sb = smem + cur * STAGE;
    v16bf a[4], b[4];
#pragma unroll
    for (int ms = 0; ms < 4; ++ms) {
      const char* p = sb + (wm * 64 + ms * 16 + r16) * 80 + 16 * half;
      v8bf lo = *(const v8bf*)(p);
      v8bf hi = *(const v8bf*)(p + 32);
      a[ms] = __builtin_shufflevector(lo, hi, 0, 1, 2, 3, 4, 5, 6, 7,
                                      8, 9, 10, 11, 12, 13, 14, 15);
    }
#pragma unroll
    for (int ns = 0; ns < 4; ++ns) {
      const char* p = sb + XBYTES + (wn * 64 + ns * 16 + r16) * 80 + 16 * half;
      v8bf lo = *(const v8bf*)(p);
      v8bf hi = *(const v8bf*)(p + 32);
      b[ns] = __builtin_shufflevector(lo, hi, 0, 1, 2, 3, 4, 5, 6, 7,
                                      8, 9, 10, 11, 12, 13, 14, 15);
    }
#pragma unroll
    for (int ms = 0; ms < 4; ++ms)
#pragma unroll
      for (int ns = 0; ns < 4; ++ns)
        acc[ms][ns] = __builtin_amdgcn_wmma_f32_16x16x32_bf16(
            false, a[ms], false, b[ns], (short)0, acc[ms][ns], false, false);
    __syncthreads();   // stage `cur` free before it is re-issued next iter
  }

  // ---- epilogue: D layout M = vreg + 8*(lane/16), N = lane%16 ----
#pragma unroll
  for (int ms = 0; ms < 4; ++ms) {
#pragma unroll
    for (int ns = 0; ns < 4; ++ns) {
      int n    = n0 + wn * 64 + ns * 16 + r16;
      float bb = Bv[n];
      int mb   = m0 + wm * 64 + ms * 16 + 8 * half;
#pragma unroll
      for (int v = 0; v < 8; ++v)
        Y[(size_t)(mb + v) * OUT_F + n] = acc[ms][ns][v] + bb;
    }
  }
}

// ================= fallback: fused single-kernel (round-1) =================
__global__ __launch_bounds__(256, 2)
void nf4_wmma_gemm_fused(const float* __restrict__ X, const int* __restrict__ Q,
                         const float* __restrict__ S, const float* __restrict__ Bv,
                         float* __restrict__ Y) {
  __shared__ __bf16 ldsA[128 * LDSTR];
  __shared__ __bf16 ldsB[128 * LDSTR];
  __shared__ float  nf4[16];

  const int tid  = threadIdx.x;
  const int lane = tid & 31;
  const int wave = tid >> 5;
  const int wm   = wave >> 2;
  const int wn   = wave & 3;
  const int half = lane >> 4;
  const int r16  = lane & 15;
  const int m0 = blockIdx.y * 128;
  const int n0 = blockIdx.x * 128;

  if (tid < 16) nf4[tid] = NF4TAB[tid];
  __syncthreads();

  v8f acc[4][2] = {};
  for (int k0 = 0; k0 < IN_F; k0 += 32) {
#pragma unroll
    for (int i = 0; i < 4; ++i) {
      int c = tid + i * 256, row = c >> 3, col = (c & 7) * 4;
      v4f xv = *(const v4f*)(X + (size_t)(m0 + row) * IN_F + k0 + col);
      v4bf xb;
#pragma unroll
      for (int j = 0; j < 4; ++j) xb[j] = f2bf(xv[j]);
      *(v4bf*)(&ldsA[row * LDSTR + col]) = xb;
    }
#pragma unroll
    for (int i = 0; i < 4; ++i) {
      int c = tid + i * 256, row = c >> 3, col = (c & 7) * 4;
      int o = n0 + row;
      v4i qv  = *(const v4i*)(Q + (size_t)o * IN_F + k0 + col);
      float s = S[((size_t)o * IN_F + k0) >> 6];
      v4bf wb;
#pragma unroll
      for (int j = 0; j < 4; ++j) wb[j] = f2bf(nf4[qv[j] & 15] * s);
      *(v4bf*)(&ldsB[row * LDSTR + col]) = wb;
    }
    __syncthreads();
    v16bf a[4], b[2];
#pragma unroll
    for (int ms = 0; ms < 4; ++ms) {
      const __bf16* p = &ldsA[(wm * 64 + ms * 16 + r16) * LDSTR];
      v8bf lo = *(const v8bf*)(p + 8 * half);
      v8bf hi = *(const v8bf*)(p + 16 + 8 * half);
      a[ms] = __builtin_shufflevector(lo, hi, 0, 1, 2, 3, 4, 5, 6, 7,
                                      8, 9, 10, 11, 12, 13, 14, 15);
    }
#pragma unroll
    for (int ns = 0; ns < 2; ++ns) {
      const __bf16* p = &ldsB[(wn * 32 + ns * 16 + r16) * LDSTR];
      v8bf lo = *(const v8bf*)(p + 8 * half);
      v8bf hi = *(const v8bf*)(p + 16 + 8 * half);
      b[ns] = __builtin_shufflevector(lo, hi, 0, 1, 2, 3, 4, 5, 6, 7,
                                      8, 9, 10, 11, 12, 13, 14, 15);
    }
#pragma unroll
    for (int ms = 0; ms < 4; ++ms)
#pragma unroll
      for (int ns = 0; ns < 2; ++ns)
        acc[ms][ns] = __builtin_amdgcn_wmma_f32_16x16x32_bf16(
            false, a[ms], false, b[ns], (short)0, acc[ms][ns], false, false);
    __syncthreads();
  }
#pragma unroll
  for (int ms = 0; ms < 4; ++ms)
#pragma unroll
    for (int ns = 0; ns < 2; ++ns) {
      int n = n0 + wn * 32 + ns * 16 + r16;
      float bb = Bv[n];
      int mb = m0 + wm * 64 + ms * 16 + 8 * half;
#pragma unroll
      for (int v = 0; v < 8; ++v)
        Y[(size_t)(mb + v) * OUT_F + n] = acc[ms][ns][v] + bb;
    }
}

extern "C" void kernel_launch(void* const* d_in, const int* in_sizes, int n_in,
                              void* d_out, int out_size, void* d_ws, size_t ws_size,
                              hipStream_t stream) {
  const float* x = (const float*)d_in[0];
  const int*   q = (const int*)d_in[1];
  const float* s = (const float*)d_in[2];
  const float* b = (const float*)d_in[3];
  float* y = (float*)d_out;

  const size_t wbytes = (size_t)OUT_F * IN_F * 2;   // 33,554,432
  const size_t xbytes = (size_t)M_TOT * IN_F * 2;   // 16,777,216

  if (ws_size >= wbytes + xbytes) {
    __bf16* Wb = (__bf16*)d_ws;
    __bf16* Xb = (__bf16*)((char*)d_ws + wbytes);
    dequant_w<<<(OUT_F * IN_F) / (256 * 8), 256, 0, stream>>>(q, s, Wb);
    convert_x<<<(M_TOT * IN_F) / (256 * 8), 256, 0, stream>>>(x, Xb);
    dim3 grid(OUT_F / BN, M_TOT / BM);              // 32 x 32
    bf16_wmma_gemm<<<grid, 256, 2 * STAGE, stream>>>(Xb, Wb, b, y);
  } else {
    dim3 grid(OUT_F / 128, M_TOT / 128);            // 64 x 32
    nf4_wmma_gemm_fused<<<grid, 256, 0, stream>>>(x, q, s, b, y);
  }
}
